// BasicQuantumAttention_9990093931329
// MI455X (gfx1250) — matrix operand order repaired and checked by reading, not compiled, not run on hardware
//
#include <hip/hip_runtime.h>
#include <hip/hip_bf16.h>

typedef __attribute__((ext_vector_type(16))) _Float16 v16h;
typedef __attribute__((ext_vector_type(8)))  _Float16 v8h;
typedef __attribute__((ext_vector_type(8)))  float    v8f;

#define B_   4
#define L_   2048
#define D_   1024
#define KT_  (L_ / 32)   // 64 k-chunks of 32
#define NT_  (D_ / 16)   // 64 n-tiles of 16
#define SCALE_ 0.03125f  // D^-0.5 for D=1024
#define LN_EPS_ 1e-5f

// ---------------------------------------------------------------------------
// Phase 1a: per-row scalar  out[row] = scale*0.01*rowsum(LayerNorm(x)*g + b)
// rowsum(LN) = rstd*(sum(g*x) - mean*sum(g)) + sum(b)
// grid = B*L blocks, 256 threads
// ---------------------------------------------------------------------------
__global__ __launch_bounds__(256) void rowscalar_kernel(
    const float* __restrict__ x, const float* __restrict__ g,
    const float* __restrict__ be, float* __restrict__ out)
{
    const int row = blockIdx.x;
    const int tid = threadIdx.x;
    const float* xr = x + (size_t)row * D_;

    float s1 = 0.f, s2 = 0.f, sgx = 0.f, sg = 0.f, sb = 0.f;
    #pragma unroll
    for (int i = 0; i < D_ / 256; ++i) {
        int d = tid + 256 * i;
        float xv = xr[d], gv = g[d], bv = be[d];
        s1 += xv; s2 += xv * xv; sgx += gv * xv; sg += gv; sb += bv;
    }
    #pragma unroll
    for (int off = 16; off; off >>= 1) {
        s1  += __shfl_xor(s1,  off);
        s2  += __shfl_xor(s2,  off);
        sgx += __shfl_xor(sgx, off);
        sg  += __shfl_xor(sg,  off);
        sb  += __shfl_xor(sb,  off);
    }
    __shared__ float red[8][5];
    const int wid = tid >> 5, lane = tid & 31;
    if (lane == 0) {
        red[wid][0] = s1; red[wid][1] = s2; red[wid][2] = sgx;
        red[wid][3] = sg; red[wid][4] = sb;
    }
    __syncthreads();
    if (tid == 0) {
        float t1 = 0, t2 = 0, t3 = 0, t4 = 0, t5 = 0;
        #pragma unroll
        for (int w = 0; w < 8; ++w) {
            t1 += red[w][0]; t2 += red[w][1]; t3 += red[w][2];
            t4 += red[w][3]; t5 += red[w][4];
        }
        const float invD = 1.0f / D_;
        float m    = t1 * invD;
        float var  = t2 * invD - m * m;
        float rstd = rsqrtf(var + LN_EPS_);
        out[row] = (rstd * (t3 - m * t4) + t5) * 0.01f * SCALE_;
    }
}

// ---------------------------------------------------------------------------
// Phase 1b: LayerNorm V, *0.01, write f16 into WMMA-B swizzled layout:
//   per (batch, kt, nt) 32x16 tile: lane = n + 16*(kk/16), half kk%16
//   flat half index = (((b*KT + kt)*NT + nt)*32 + lane)*16 + (kk&15)
// grid = B*L blocks, 256 threads
// ---------------------------------------------------------------------------
__global__ __launch_bounds__(256) void vswizzle_kernel(
    const float* __restrict__ x, const float* __restrict__ g,
    const float* __restrict__ be, _Float16* __restrict__ dst)
{
    const int row = blockIdx.x;          // b*L + jj
    const int tid = threadIdx.x;
    const float* xr = x + (size_t)row * D_;

    float xv[D_ / 256];
    float s1 = 0.f, s2 = 0.f;
    #pragma unroll
    for (int i = 0; i < D_ / 256; ++i) {
        xv[i] = xr[tid + 256 * i];
        s1 += xv[i]; s2 += xv[i] * xv[i];
    }
    #pragma unroll
    for (int off = 16; off; off >>= 1) {
        s1 += __shfl_xor(s1, off);
        s2 += __shfl_xor(s2, off);
    }
    __shared__ float red1[8], red2[8];
    const int wid = tid >> 5, lane = tid & 31;
    if (lane == 0) { red1[wid] = s1; red2[wid] = s2; }
    __syncthreads();
    float t1 = 0, t2 = 0;
    #pragma unroll
    for (int w = 0; w < 8; ++w) { t1 += red1[w]; t2 += red2[w]; }
    const float invD = 1.0f / D_;
    float m    = t1 * invD;
    float rstd = rsqrtf(t2 * invD - m * m + LN_EPS_);

    const int b   = row >> 11;
    const int jj  = row & (L_ - 1);
    const int kt  = jj >> 5;
    const int kk  = jj & 31;
    const int lg  = kk >> 4;      // lane group
    const int kk2 = kk & 15;      // half slot within lane
    #pragma unroll
    for (int i = 0; i < D_ / 256; ++i) {
        int d = tid + 256 * i;
        float y = ((xv[i] - m) * rstd * g[d] + be[d]) * 0.01f;
        int nt = d >> 4, n = d & 15;
        size_t idx = ((((size_t)b * KT_ + kt) * NT_ + nt) * 32 + (n + 16 * lg)) * 16 + kk2;
        dst[idx] = (_Float16)y;
    }
}

// ---------------------------------------------------------------------------
// Phase 2: attention + GEMM + fused output LayerNorm.
// grid = B * (L/32) blocks, 256 threads = 8 waves.
// Block owns a 32-row i-tile over all D=1024; each wave keeps TWO 16-row
// accumulator sets so every B tile load feeds 4 WMMAs (32 FLOP / L2 byte).
// ---------------------------------------------------------------------------
__global__ __launch_bounds__(256) void attn_kernel(
    const float* __restrict__ Ar, const float* __restrict__ Ai,
    const float* __restrict__ Br, const float* __restrict__ Bi,
    const _Float16* __restrict__ Vr, const _Float16* __restrict__ Vi,
    const float* __restrict__ og, const float* __restrict__ ob,
    float* __restrict__ out)
{
    __shared__ float Zl[32];
    __shared__ __align__(32) _Float16 Pl[32 * 32];
    __shared__ float Sr[32], Sr2[32], Si[32], Si2[32];

    const int b  = blockIdx.x >> 6;          // 64 i-tiles of 32 per batch
    const int i0 = (blockIdx.x & 63) << 5;
    const int tid  = threadIdx.x;
    const int wid  = tid >> 5;
    const int lane = tid & 31;

    const float* ArB = Ar + b * L_ + i0;
    const float* AiB = Ai + b * L_ + i0;
    const float* BrB = Br + b * L_;
    const float* BiB = Bi + b * L_;

    // ---- softmax denominators (logits bounded in [-10,10], no max needed) ----
    if (tid < 32) Zl[tid] = 0.f;
    __syncthreads();
    {
        int r  = tid & 31;
        int j0 = tid >> 5;
        float ar = ArB[r] + 1e-8f, ai = AiB[r] + 1e-8f;
        float z = 0.f;
        for (int j = j0; j < L_; j += 8) {
            float xx = ar - BrB[j];
            float yy = ai - BiB[j];
            float s  = 10.f * xx * rsqrtf(xx * xx + yy * yy);
            z += __expf(s);
        }
        atomicAdd(&Zl[r], z);
    }
    __syncthreads();

    // constants for this thread's four P entries per chunk (row pr, k = pk..pk+3)
    const int   pr  = tid >> 3;
    const int   pk  = (tid & 7) * 4;
    const float par  = ArB[pr] + 1e-8f;
    const float pai  = AiB[pr] + 1e-8f;
    const float invZ = 1.0f / Zl[pr];

    v8f accR0[8], accI0[8], accR1[8], accI1[8];
    #pragma unroll
    for (int t = 0; t < 8; ++t) {
        accR0[t] = {}; accI0[t] = {}; accR1[t] = {}; accI1[t] = {};
    }

    const int m     = lane & 15;
    const int abase = (lane < 16) ? 0 : 16;
    const v16h* vrv = reinterpret_cast<const v16h*>(Vr);
    const v16h* viv = reinterpret_cast<const v16h*>(Vi);

    for (int kt = 0; kt < KT_; ++kt) {
        // compute 32x32 probability chunk into LDS (row-major halves)
        {
            int j = kt * 32 + pk;
            #pragma unroll
            for (int u = 0; u < 4; ++u) {
                float xx = par - BrB[j + u];
                float yy = pai - BiB[j + u];
                float s  = 10.f * xx * rsqrtf(xx * xx + yy * yy);
                Pl[pr * 32 + pk + u] = (_Float16)(__expf(s) * invZ);
            }
        }
        __syncthreads();

        // A operands (two 16-row tiles), contiguous k = 16*(lane/16) + s, same order as B
        v16h Aop0, Aop1;
        {
            const v8h* p0 = reinterpret_cast<const v8h*>(&Pl[m * 32 + abase]);
            const v8h* p1 = reinterpret_cast<const v8h*>(&Pl[(m + 16) * 32 + abase]);
            v8h lo0 = p0[0], hi0 = p0[1];   // k = abase..+7, abase+8..+15
            v8h lo1 = p1[0], hi1 = p1[1];
            #pragma unroll
            for (int i = 0; i < 8; ++i) {
                Aop0[i] = lo0[i]; Aop0[i + 8] = hi0[i];
                Aop1[i] = lo1[i]; Aop1[i + 8] = hi1[i];
            }
        }

        // 8 n-tiles per wave; each B tile (coalesced b128 pair) feeds 4 WMMAs
        size_t tbase = (((size_t)b * KT_ + kt) * NT_ + wid * 8) * 32 + lane;
        #pragma unroll
        for (int t = 0; t < 8; ++t) {
            v16h Bvr = vrv[tbase + (size_t)t * 32];
            v16h Bvi = viv[tbase + (size_t)t * 32];
            accR0[t] = __builtin_amdgcn_wmma_f32_16x16x32_f16(
                false, Aop0, false, Bvr, (short)0, accR0[t], false, false);
            accR1[t] = __builtin_amdgcn_wmma_f32_16x16x32_f16(
                false, Aop1, false, Bvr, (short)0, accR1[t], false, false);
            accI0[t] = __builtin_amdgcn_wmma_f32_16x16x32_f16(
                false, Aop0, false, Bvi, (short)0, accI0[t], false, false);
            accI1[t] = __builtin_amdgcn_wmma_f32_16x16x32_f16(
                false, Aop1, false, Bvi, (short)0, accI1[t], false, false);
        }
        __syncthreads();   // safe to overwrite Pl next iteration
    }

    // ---- fused output LayerNorm (mean/var over D across the 8 waves) ----
    if (tid < 32) { Sr[tid] = 0.f; Sr2[tid] = 0.f; Si[tid] = 0.f; Si2[tid] = 0.f; }
    __syncthreads();

    #pragma unroll
    for (int a = 0; a < 8; ++a) {
        float sr0 = 0, sr20 = 0, si0 = 0, si20 = 0;
        float sr1 = 0, sr21 = 0, si1 = 0, si21 = 0;
        #pragma unroll
        for (int t = 0; t < 8; ++t) {
            float vr0 = accR0[t][a] * 0.1f, vi0 = accI0[t][a] * 0.1f;
            float vr1 = accR1[t][a] * 0.1f, vi1 = accI1[t][a] * 0.1f;
            sr0 += vr0; sr20 += vr0 * vr0; si0 += vi0; si20 += vi0 * vi0;
            sr1 += vr1; sr21 += vr1 * vr1; si1 += vi1; si21 += vi1 * vi1;
        }
        #pragma unroll
        for (int off = 8; off; off >>= 1) {   // reduce 16 columns per half-wave
            sr0  += __shfl_xor(sr0,  off); sr20 += __shfl_xor(sr20, off);
            si0  += __shfl_xor(si0,  off); si20 += __shfl_xor(si20, off);
            sr1  += __shfl_xor(sr1,  off); sr21 += __shfl_xor(sr21, off);
            si1  += __shfl_xor(si1,  off); si21 += __shfl_xor(si21, off);
        }
        if ((lane & 15) == 0) {
            int row = ((lane < 16) ? a : (a + 8));
            atomicAdd(&Sr[row], sr0);       atomicAdd(&Sr2[row], sr20);
            atomicAdd(&Si[row], si0);       atomicAdd(&Si2[row], si20);
            atomicAdd(&Sr[row + 16], sr1);  atomicAdd(&Sr2[row + 16], sr21);
            atomicAdd(&Si[row + 16], si1);  atomicAdd(&Si2[row + 16], si21);
        }
    }
    __syncthreads();

    const float invD = 1.0f / D_;
    const int   n    = lane & 15;
    float og_[8], ob_[8];
    #pragma unroll
    for (int t = 0; t < 8; ++t) {
        int d = wid * 128 + t * 16 + n;
        og_[t] = og[d]; ob_[t] = ob[d];
    }
    float* outR = out;
    float* outI = out + (size_t)B_ * L_ * D_;

    #pragma unroll
    for (int a = 0; a < 8; ++a) {
        int row0 = ((lane < 16) ? a : (a + 8));
        #pragma unroll
        for (int half = 0; half < 2; ++half) {
            int row = row0 + 16 * half;
            float mr = Sr[row] * invD;
            float rr = rsqrtf(fmaxf(Sr2[row] * invD - mr * mr, 0.f) + LN_EPS_);
            float mi = Si[row] * invD;
            float ri = rsqrtf(fmaxf(Si2[row] * invD - mi * mi, 0.f) + LN_EPS_);
            size_t rowoff = ((size_t)b * L_ + i0 + row) * D_;
            #pragma unroll
            for (int t = 0; t < 8; ++t) {
                int d = wid * 128 + t * 16 + n;
                float vr_ = (half ? accR1[t][a] : accR0[t][a]) * 0.1f;
                float vi_ = (half ? accI1[t][a] : accI0[t][a]) * 0.1f;
                outR[rowoff + d] = (vr_ - mr) * rr * og_[t] + ob_[t];
                outI[rowoff + d] = (vi_ - mi) * ri * og_[t] + ob_[t];
            }
        }
    }
}

// ---------------------------------------------------------------------------
extern "C" void kernel_launch(void* const* d_in, const int* in_sizes, int n_in,
                              void* d_out, int out_size, void* d_ws, size_t ws_size,
                              hipStream_t stream)
{
    const float* q_real = (const float*)d_in[0];
    const float* q_imag = (const float*)d_in[1];
    const float* k_real = (const float*)d_in[2];
    const float* k_imag = (const float*)d_in[3];
    const float* v_real = (const float*)d_in[4];
    const float* v_imag = (const float*)d_in[5];
    const float* q_gamma = (const float*)d_in[6];
    const float* q_beta  = (const float*)d_in[7];
    const float* k_gamma = (const float*)d_in[8];
    const float* k_beta  = (const float*)d_in[9];
    const float* v_gamma = (const float*)d_in[10];
    const float* v_beta  = (const float*)d_in[11];
    const float* out_gamma = (const float*)d_in[12];
    const float* out_beta  = (const float*)d_in[13];

    const int nrows = B_ * L_;                 // 8192
    float* ws = (float*)d_ws;
    float* Ar = ws;                            // q_real row scalars
    float* Ai = ws + nrows;                    // q_imag
    float* Brr = ws + 2 * nrows;               // k_real
    float* Bii = ws + 3 * nrows;               // k_imag
    _Float16* Vr = (_Float16*)((char*)d_ws + 4 * (size_t)nrows * sizeof(float));
    _Float16* Vi = Vr + (size_t)B_ * L_ * D_;

    rowscalar_kernel<<<nrows, 256, 0, stream>>>(q_real, q_gamma, q_beta, Ar);
    rowscalar_kernel<<<nrows, 256, 0, stream>>>(q_imag, q_gamma, q_beta, Ai);
    rowscalar_kernel<<<nrows, 256, 0, stream>>>(k_real, k_gamma, k_beta, Brr);
    rowscalar_kernel<<<nrows, 256, 0, stream>>>(k_imag, k_gamma, k_beta, Bii);
    vswizzle_kernel<<<nrows, 256, 0, stream>>>(v_real, v_gamma, v_beta, Vr);
    vswizzle_kernel<<<nrows, 256, 0, stream>>>(v_imag, v_gamma, v_beta, Vi);

    attn_kernel<<<B_ * (L_ / 32), 256, 0, stream>>>(
        Ar, Ai, Brr, Bii, Vr, Vi, out_gamma, out_beta, (float*)d_out);
}
